// PositionEmbedding_54958401519805
// MI455X (gfx1250) — matrix operand order repaired and checked by reading, not compiled, not run on hardware
//
#include <hip/hip_runtime.h>
#include <math.h>

// ---------------------------------------------------------------------------
// Causal attention with column-softmax (softmax over axis=1), bf16 WMMA path.
// MI455X / gfx1250: wave32, V_WMMA_F32_16X16X32_BF16, fp32 accumulation.
// x is pre-converted to bf16 once so every GEMM k-loop is pure load+WMMA.
// ---------------------------------------------------------------------------

typedef __attribute__((ext_vector_type(16))) __bf16 v16bf;
typedef __attribute__((ext_vector_type(8)))  __bf16 v8bf;
typedef __attribute__((ext_vector_type(8)))  float  v8f;

static constexpr int Bn  = 64;
static constexpr int T   = 512;
static constexpr int E   = 512;
static constexpr int Kd  = 512;
static constexpr int Vd  = 512;
static constexpr int BCH = 8;                 // batch chunk for scores buffer

union V16u { v16bf v; v8bf h[2]; __bf16 e[16]; unsigned u[8]; };
union C8u  { v8f  v; float e[8]; };
union P8u  { v8bf v; unsigned u[4]; };

// ----------------------- fp32 -> bf16 (RNE) helpers ------------------------
__device__ __forceinline__ unsigned bfbits(float f) {
  unsigned u = __builtin_bit_cast(unsigned, f);
  u += 0x7FFFu + ((u >> 16) & 1u);
  return u >> 16;
}
__device__ __forceinline__ __bf16 f2bf(float f) {
  unsigned short h = (unsigned short)bfbits(f);
  return __builtin_bit_cast(__bf16, h);
}
// packed pair: lo -> [15:0], hi -> [31:16]
__device__ __forceinline__ unsigned pkbf(float lo, float hi) {
  return bfbits(lo) | (bfbits(hi) << 16);
}

#define WMMA_BF16(A, B, C) \
  __builtin_amdgcn_wmma_f32_16x16x32_bf16(false, (A), false, (B), (short)0, (C), false, false)

// --------------------------- x convert (once) ------------------------------
// x (B*T,E) fp32 -> Xb bf16 row-major. 8 elements/thread, 128-bit stores.
__global__ __launch_bounds__(256) void xconv_kernel(
    const float* __restrict__ x, __bf16* __restrict__ Xb) {
  size_t i = ((size_t)blockIdx.x * 256 + threadIdx.x) * 8;
  v8f f = *(const v8f*)(x + i);
  P8u p;
#pragma unroll
  for (int e = 0; e < 4; ++e) p.u[e] = pkbf(f[2 * e], f[2 * e + 1]);
  *(v8bf*)(Xb + i) = p.v;
}

// --------------------------- weight transpose+convert ----------------------
// W (E,Kd) fp32 row-major  ->  Wt (Kd,E) bf16 row-major, so the WMMA B-operand
// (lane n reads two runs of 8 contiguous k-elements, 16 apart) is a pair of 128-bit loads.
__global__ __launch_bounds__(256) void wconv_kernel(
    const float* __restrict__ Wq, const float* __restrict__ Wk,
    const float* __restrict__ Wv,
    __bf16* __restrict__ Wqt, __bf16* __restrict__ Wkt, __bf16* __restrict__ Wvt) {
  int idx = blockIdx.x * 256 + threadIdx.x;   // E*Kd threads
  int e = idx / Kd;
  int n = idx % Kd;
  size_t src = (size_t)e * Kd + n;
  size_t dst = (size_t)n * E + e;
  Wqt[dst] = f2bf(Wq[src]);
  Wkt[dst] = f2bf(Wk[src]);
  Wvt[dst] = f2bf(Wv[src]);
}

// ------------------------------- QKV projection ----------------------------
// One wave per 32x16 output tile, shared across the three projections:
// 6 WMMAs per k-step (2 A frags x {Wq,Wk,Wv}). Pure load+WMMA inner loop.
// Q,K stored row-major bf16; V stored transposed per batch (Vt[v][t]).
__global__ __launch_bounds__(32) void qkv_proj_kernel(
    const __bf16* __restrict__ Xb,
    const __bf16* __restrict__ Wqt, const __bf16* __restrict__ Wkt,
    const __bf16* __restrict__ Wvt,
    const float* __restrict__ bq, const float* __restrict__ bk,
    const float* __restrict__ bv,
    __bf16* __restrict__ Qb, __bf16* __restrict__ Kb, __bf16* __restrict__ Vt) {
  int tile = blockIdx.x;
  int ct = tile & 31;           // N tile (32 per 512)
  int rt = tile >> 5;           // 32-row tile over B*T (1024 tiles)
  int lane = threadIdx.x;
  int ml = lane & 15;
  int mh = lane >> 4;           // lane half
  int n0 = ct * 16;

  const __bf16* x0 = Xb + (size_t)(rt * 32 + ml) * E;   // rows rt*32 .. +15
  const __bf16* x1 = x0 + (size_t)16 * E;               // rows +16 .. +31
  const __bf16* wq = Wqt + (size_t)(n0 + ml) * E;
  const __bf16* wk = Wkt + (size_t)(n0 + ml) * E;
  const __bf16* wv = Wvt + (size_t)(n0 + ml) * E;

  v8f cq0 = {}, cq1 = {}, ck0 = {}, ck1 = {}, cv0 = {}, cv1 = {};
  for (int kk = 0; kk < E; kk += 32) {
    __builtin_prefetch(x0 + kk + 64, 0, 1);
    V16u a0, a1, fq, fk, fv;
    a0.h[0] = *(const v8bf*)(x0 + kk + mh * 8);
    a0.h[1] = *(const v8bf*)(x0 + kk + 16 + mh * 8);
    a1.h[0] = *(const v8bf*)(x1 + kk + mh * 8);
    a1.h[1] = *(const v8bf*)(x1 + kk + 16 + mh * 8);
    int kb = kk + mh * 8;
    fq.h[0] = *(const v8bf*)(wq + kb); fq.h[1] = *(const v8bf*)(wq + kb + 16);
    fk.h[0] = *(const v8bf*)(wk + kb); fk.h[1] = *(const v8bf*)(wk + kb + 16);
    fv.h[0] = *(const v8bf*)(wv + kb); fv.h[1] = *(const v8bf*)(wv + kb + 16);

    cq0 = WMMA_BF16(a0.v, fq.v, cq0);
    cq1 = WMMA_BF16(a1.v, fq.v, cq1);
    ck0 = WMMA_BF16(a0.v, fk.v, ck0);
    ck1 = WMMA_BF16(a1.v, fk.v, ck1);
    cv0 = WMMA_BF16(a0.v, fv.v, cv0);
    cv1 = WMMA_BF16(a1.v, fv.v, cv1);
  }

  int n = n0 + ml;              // C/D: lane holds column n, rows r+8*mh
  float biq = bq[n], bik = bk[n], biv = bv[n];
  C8u q0, q1, k0, k1, v0, v1;
  q0.v = cq0; q1.v = cq1; k0.v = ck0; k1.v = ck1; v0.v = cv0; v1.v = cv1;
  int batch = (rt * 32) / T;    // 32-row tiles never straddle batches (T%32==0)
#pragma unroll
  for (int r = 0; r < 8; ++r) {
    int rr = rt * 32 + r + 8 * mh;
    Qb[(size_t)rr * Kd + n]        = f2bf(q0.e[r] + biq);
    Qb[(size_t)(rr + 16) * Kd + n] = f2bf(q1.e[r] + biq);
    Kb[(size_t)rr * Kd + n]        = f2bf(k0.e[r] + bik);
    Kb[(size_t)(rr + 16) * Kd + n] = f2bf(k1.e[r] + bik);
  }
  // V transposed: 8 consecutive t per lane -> 128-bit stores
  int tloc = rt * 32 + 8 * mh - batch * T;
  __bf16* vp = Vt + (size_t)batch * Vd * T + (size_t)n * T;
  P8u vv0, vv1;
#pragma unroll
  for (int e = 0; e < 4; ++e) {
    vv0.u[e] = pkbf(v0.e[2 * e] + biv, v0.e[2 * e + 1] + biv);
    vv1.u[e] = pkbf(v1.e[2 * e] + biv, v1.e[2 * e + 1] + biv);
  }
  *(v8bf*)(vp + tloc)      = vv0.v;
  *(v8bf*)(vp + tloc + 16) = vv1.v;
}

// ---------------------- scores^T = (Q K^T)^T, 32x32/wave -------------------
// Stores St[s][t] (transposed) so the axis=1 softmax reduces over contiguous
// memory and D-tile stores are contiguous v8f. Fully masked 32x32 tiles
// (s0 >= t0+32) are skipped.
__global__ __launch_bounds__(32) void scores_kernel(
    const __bf16* __restrict__ Qb, const __bf16* __restrict__ Kb,
    float* __restrict__ St, int b0) {
  int blk = blockIdx.x;
  int st = blk & 15;
  int tt = (blk >> 4) & 15;
  int bl = blk >> 8;            // 0..BCH-1
  int t0 = tt * 32, s0 = st * 32;
  if (s0 >= t0 + 32) return;    // uniform per wave: tile entirely above diagonal
  int b = b0 + bl;
  int lane = threadIdx.x;
  int ml = lane & 15;
  int mh = lane >> 4;

  const __bf16* q0 = Qb + ((size_t)b * T + t0 + ml) * Kd;       // t rows
  const __bf16* q1 = q0 + (size_t)16 * Kd;
  const __bf16* k0 = Kb + ((size_t)b * T + s0 + ml) * Kd;       // s rows
  const __bf16* k1 = k0 + (size_t)16 * Kd;

  v8f c00 = {}, c01 = {}, c10 = {}, c11 = {};
  for (int kk = 0; kk < Kd; kk += 32) {
    __builtin_prefetch(q0 + kk + 64, 0, 1);
    __builtin_prefetch(k0 + kk + 64, 0, 1);
    V16u a0, a1, f0, f1;
    a0.h[0] = *(const v8bf*)(q0 + kk + mh * 8);
    a0.h[1] = *(const v8bf*)(q0 + kk + 16 + mh * 8);
    a1.h[0] = *(const v8bf*)(q1 + kk + mh * 8);
    a1.h[1] = *(const v8bf*)(q1 + kk + 16 + mh * 8);
    f0.h[0] = *(const v8bf*)(k0 + kk + mh * 8);
    f0.h[1] = *(const v8bf*)(k0 + kk + 16 + mh * 8);
    f1.h[0] = *(const v8bf*)(k1 + kk + mh * 8);
    f1.h[1] = *(const v8bf*)(k1 + kk + 16 + mh * 8);
    c00 = WMMA_BF16(a0.v, f0.v, c00);
    c01 = WMMA_BF16(a0.v, f1.v, c01);
    c10 = WMMA_BF16(a1.v, f0.v, c10);
    c11 = WMMA_BF16(a1.v, f1.v, c11);
  }
  // D tile (m=t, n=s) -> St[s][t]: per-lane s fixed, 8 consecutive t.
  float* Sp = St + (size_t)bl * T * T;
  int tb = t0 + 8 * mh;
  *(v8f*)(Sp + (size_t)(s0 + ml) * T + tb)            = c00;
  *(v8f*)(Sp + (size_t)(s0 + 16 + ml) * T + tb)       = c01;
  *(v8f*)(Sp + (size_t)(s0 + ml) * T + tb + 16)       = c10;
  *(v8f*)(Sp + (size_t)(s0 + 16 + ml) * T + tb + 16)  = c11;
}

// -------------------------- column softmax (axis=1) ------------------------
// One 256-thread block per (batch-in-chunk, column s). Scores are stored
// transposed, so this reduces over a contiguous row. Valid rows are t >= s.
// Writes the FULL attn column (zeros for t < s) so the downstream GEMM can
// read any k-block without touching uninitialized workspace.
__global__ __launch_bounds__(256) void colsoftmax_kernel(
    const float* __restrict__ St, __bf16* __restrict__ At) {
  int blk = blockIdx.x;
  int s  = blk & (T - 1);
  int bl = blk / T;
  const float* Sp = St + (size_t)bl * T * T + (size_t)s * T;  // contiguous in t
  __bf16* Ap = At + (size_t)bl * T * T;
  int tid = threadIdx.x;
  const float scale = 0.044194173824159216f;   // 1/sqrt(512)

  __shared__ float red[256];
  float m = -INFINITY;
  for (int t = tid; t < T; t += 256)
    if (t >= s) m = fmaxf(m, Sp[t] * scale);
  red[tid] = m; __syncthreads();
  for (int off = 128; off > 0; off >>= 1) {
    if (tid < off) red[tid] = fmaxf(red[tid], red[tid + off]);
    __syncthreads();
  }
  m = red[0]; __syncthreads();

  float sum = 0.f;
  for (int t = tid; t < T; t += 256)
    if (t >= s) sum += __expf(Sp[t] * scale - m);
  red[tid] = sum; __syncthreads();
  for (int off = 128; off > 0; off >>= 1) {
    if (tid < off) red[tid] += red[tid + off];
    __syncthreads();
  }
  float inv = 1.f / red[0];

  for (int t = tid; t < T; t += 256) {
    float p = 0.f;
    if (t >= s) p = __expf(Sp[t] * scale - m) * inv;
    Ap[(size_t)t * T + s] = f2bf(p);     // attn row-major [t][s]
  }
}

// ------------------- out^T = V^T attn^T, 32x32 per wave --------------------
// A = Vt rows (v, contiguous s); B = At rows (t, contiguous s); D tile is
// (m=v, n=t) so each lane stores 8 CONSECUTIVE v elements of out[t][:].
// k-loop truncated at s < t0+32 (attn above diagonal is exactly zero).
__global__ __launch_bounds__(32) void outgemm_kernel(
    const __bf16* __restrict__ At, const __bf16* __restrict__ Vt,
    float* __restrict__ out, int b0) {
  int blk = blockIdx.x;
  int vt = blk & 15;
  int tt = (blk >> 4) & 15;
  int bl = blk >> 8;
  int b = b0 + bl;
  int v0 = vt * 32, t0 = tt * 32;
  int lane = threadIdx.x;
  int ml = lane & 15;
  int mh = lane >> 4;

  const __bf16* a0 = Vt + (size_t)b * Vd * T + (size_t)(v0 + ml) * T;   // v rows
  const __bf16* a1 = a0 + (size_t)16 * T;
  const __bf16* p0 = At + (size_t)bl * T * T + (size_t)(t0 + ml) * T;   // t rows
  const __bf16* p1 = p0 + (size_t)16 * T;

  v8f c00 = {}, c01 = {}, c10 = {}, c11 = {};
  int kend = t0 + 32;            // s >= t0+32 contributes exact zeros
  for (int kk = 0; kk < kend; kk += 32) {
    __builtin_prefetch(a0 + kk + 64, 0, 1);
    __builtin_prefetch(p0 + kk + 64, 0, 1);
    V16u A0, A1, B0, B1;
    A0.h[0] = *(const v8bf*)(a0 + kk + mh * 8);
    A0.h[1] = *(const v8bf*)(a0 + kk + 16 + mh * 8);
    A1.h[0] = *(const v8bf*)(a1 + kk + mh * 8);
    A1.h[1] = *(const v8bf*)(a1 + kk + 16 + mh * 8);
    B0.h[0] = *(const v8bf*)(p0 + kk + mh * 8);
    B0.h[1] = *(const v8bf*)(p0 + kk + 16 + mh * 8);
    B1.h[0] = *(const v8bf*)(p1 + kk + mh * 8);
    B1.h[1] = *(const v8bf*)(p1 + kk + 16 + mh * 8);
    c00 = WMMA_BF16(A0.v, B0.v, c00);
    c01 = WMMA_BF16(A0.v, B1.v, c01);
    c10 = WMMA_BF16(A1.v, B0.v, c10);
    c11 = WMMA_BF16(A1.v, B1.v, c11);
  }
  // D tile (m=v, n=t) -> out[t][v]: per-lane t fixed, 8 consecutive v.
  float* op = out + (size_t)b * T * Vd;
  int vb = v0 + 8 * mh;
  *(v8f*)(op + (size_t)(t0 + ml) * Vd + vb)           = c00;
  *(v8f*)(op + (size_t)(t0 + 16 + ml) * Vd + vb)      = c01;
  *(v8f*)(op + (size_t)(t0 + ml) * Vd + vb + 16)      = c10;
  *(v8f*)(op + (size_t)(t0 + 16 + ml) * Vd + vb + 16) = c11;
}

// --------------------------------- launch ----------------------------------
extern "C" void kernel_launch(void* const* d_in, const int* in_sizes, int n_in,
                              void* d_out, int out_size, void* d_ws, size_t ws_size,
                              hipStream_t stream) {
  const float* x  = (const float*)d_in[0];
  const float* Wq = (const float*)d_in[1];
  const float* bq = (const float*)d_in[2];
  const float* Wk = (const float*)d_in[3];
  const float* bk = (const float*)d_in[4];
  const float* Wv = (const float*)d_in[5];
  const float* bv = (const float*)d_in[6];
  float* out = (float*)d_out;

  char* ws = (char*)d_ws;
  size_t off = 0;
  auto alloc = [&](size_t bytes) -> void* {
    void* p = ws + off;
    off = (off + bytes + 255) & ~(size_t)255;
    return p;
  };
  __bf16* Xb  = (__bf16*)alloc((size_t)Bn * T * E * 2);
  __bf16* Wqt = (__bf16*)alloc((size_t)E * Kd * 2);
  __bf16* Wkt = (__bf16*)alloc((size_t)E * Kd * 2);
  __bf16* Wvt = (__bf16*)alloc((size_t)E * Kd * 2);
  __bf16* Qb  = (__bf16*)alloc((size_t)Bn * T * Kd * 2);
  __bf16* Kb  = (__bf16*)alloc((size_t)Bn * T * Kd * 2);
  __bf16* Vt  = (__bf16*)alloc((size_t)Bn * Vd * T * 2);
  float*  St  = (float*) alloc((size_t)BCH * T * T * 4);
  __bf16* At  = (__bf16*)alloc((size_t)BCH * T * T * 2);

  // 1) one-shot conversions: x -> bf16, weights -> transposed bf16
  xconv_kernel<<<(int)((size_t)Bn * T * E / 8 / 256), 256, 0, stream>>>(x, Xb);
  wconv_kernel<<<(E * Kd) / 256, 256, 0, stream>>>(Wq, Wk, Wv, Wqt, Wkt, Wvt);

  // 2) fused QKV projection (2 A frags shared across 3 weights, 6 WMMA/step)
  qkv_proj_kernel<<<(Bn * T / 32) * (Kd / 16), 32, 0, stream>>>(
      Xb, Wqt, Wkt, Wvt, bq, bk, bv, Qb, Kb, Vt);

  // 3) per-chunk: scores^T -> column softmax -> attn @ V (as out^T = V^T P^T)
  const int tiles = BCH * (T / 32) * (T / 32);
  for (int b0 = 0; b0 < Bn; b0 += BCH) {
    scores_kernel<<<tiles, 32, 0, stream>>>(Qb, Kb, St, b0);
    colsoftmax_kernel<<<BCH * T, 256, 0, stream>>>(St, At);
    outgemm_kernel<<<tiles, 32, 0, stream>>>(At, Vt, out, b0);
  }
}
